// RoPEAttentionBNC_83399674953964
// MI455X (gfx1250) — matrix operand order repaired and checked by reading, not compiled, not run on hardware
//
#include <hip/hip_runtime.h>
#include <hip/hip_bf16.h>

typedef __attribute__((ext_vector_type(16))) __bf16 v16bf;
typedef __attribute__((ext_vector_type(8)))  float  v8f;

constexpr int CDIM  = 1024;
constexpr int NHEAD = 16;
constexpr int HD    = 64;
constexpr int BATCH = 2;
constexpr int NQ    = 2048;
constexpr int NKTOT = 2112;

union AFrag { v16bf v; unsigned int u[8]; uint4 q4[2]; };

__device__ __forceinline__ unsigned short f2bf(float f) {
  unsigned int u = __float_as_uint(f);
  u += 0x7fffu + ((u >> 16) & 1u);          // round-to-nearest-even
  return (unsigned short)(u >> 16);
}
__device__ __forceinline__ unsigned int packbf2(float a, float b) {
  return (unsigned int)f2bf(a) | ((unsigned int)f2bf(b) << 16);
}

__device__ __forceinline__ v8f wmma_bf16(v16bf a, v16bf b, v8f c) {
  return __builtin_amdgcn_wmma_f32_16x16x32_bf16(false, a, false, b, (short)0, c,
                                                 false, false);
}

// CDNA5 async memory->LDS copy (ASYNCcnt path).
__device__ __forceinline__ void async_load_b128(unsigned lds_addr,
                                                const void* gaddr) {
  asm volatile("global_load_async_to_lds_b128 %0, %1, off"
               :: "v"(lds_addr), "v"(gaddr) : "memory");
}
__device__ __forceinline__ void wait_asynccnt0() {
  asm volatile("s_wait_asynccnt 0" ::: "memory");
}

// A-fragment (16x32 bf16, M x K): lane m = lane&15, hi = lane>>4.
// slot j -> k = 16*hi + j; pairs contiguous -> b128-able loads.
template <typename P>
__device__ __forceinline__ v16bf load_a_frag(P base, int ld) {
  const int lane = threadIdx.x & 31;
  const int m = lane & 15, hi = lane >> 4;
  AFrag f;
  P row = base + m * ld + hi * 16;
#pragma unroll
  for (int v2 = 0; v2 < 4; ++v2) {
    f.u[v2]     = *(const unsigned int*)(row + 2 * v2);
    f.u[v2 + 4] = *(const unsigned int*)(row + 8 + 2 * v2);
  }
  return f.v;
}

// B-fragment (32x16 bf16, K x N) from "BT" storage: element (k,n) at
// base[n*ld + k]. slot j -> k = 16*hi + j -> 32 contiguous bytes per lane.
template <typename P>
__device__ __forceinline__ v16bf load_b_frag(P base, int ld) {
  const int lane = threadIdx.x & 31;
  const int n = lane & 15, hi = lane >> 4;
  AFrag f;
  const unsigned short* row = (const unsigned short*)(base + n * ld + hi * 16);
  f.q4[0] = *(const uint4*)(row);
  f.q4[1] = *(const uint4*)(row + 8);
  return f.v;
}

// ---------------- prep kernels ----------------

__global__ void cvt_bf16_kernel(const float* __restrict__ src,
                                unsigned short* __restrict__ dst, int n2) {
  int i = blockIdx.x * blockDim.x + threadIdx.x;
  if (i < n2) {
    float2 f = ((const float2*)src)[i];
    ((unsigned int*)dst)[i] = packbf2(f.x, f.y);
  }
}

// W is [c_in][c_out] fp32 (x @ W); produce WT [c_out][c_in] bf16 via LDS tiles.
__global__ __launch_bounds__(256) void wt_cvt_kernel(
    const float* __restrict__ W, unsigned short* __restrict__ WT) {
  __shared__ float tile[32][33];
  const int co = blockIdx.x * 32 + threadIdx.x;
  const int ci0 = blockIdx.y * 32;
#pragma unroll
  for (int i = threadIdx.y; i < 32; i += 8)
    tile[i][threadIdx.x] = W[(size_t)(ci0 + i) * CDIM + co];
  __syncthreads();
#pragma unroll
  for (int i = threadIdx.y; i < 32; i += 8)
    WT[(size_t)(blockIdx.x * 32 + i) * CDIM + ci0 + threadIdx.x] =
        f2bf(tile[threadIdx.x][i]);
}

// SAM axial RoPE tables: 2048 tokens x 32 pairs.
__global__ void rope_table_kernel(float* __restrict__ rc, float* __restrict__ rs,
                                  const int* __restrict__ qw_p) {
  int id = blockIdx.x * blockDim.x + threadIdx.x;
  if (id >= NQ * 32) return;
  int n = id >> 5, i = id & 31;
  int W = qw_p[0];
  float pos = (i < 16) ? (float)(n % W) : (float)(n / W);
  int j = i & 15;
  float freq = powf(10000.f, -(4.f * (float)j) / 64.f);
  float ang = pos * freq;
  rc[id] = cosf(ang);
  rs[id] = sinf(ang);
}

// ---------------- fused bf16 GEMM (projections + epilogues) ----------------
// Block = 128 threads (4 waves). Block tile 128x64; wave tile 32x64.
// modes: 0 = Q proj (+bias,+RoPE, heads bf16)   out_h[b][h][n][64]
//        1 = K proj (+bias, RoPE for n<n_keep)  out_h[b][h][n][64]
//        2 = V proj (+bias, transposed bf16)    out_h[b][h][dv][NKTOT]
//        3 = out proj (+bias, fp32)             out_f[row][C]
__global__ __launch_bounds__(128) void gemm_bf16_kernel(
    const unsigned short* __restrict__ X,    // [M][C] bf16
    const unsigned short* __restrict__ WT,   // [C][C] bf16, row = c_out
    const float* __restrict__ bias,          // [C]
    const float* __restrict__ ropeC, const float* __restrict__ ropeS,
    unsigned short* __restrict__ out_h, float* __restrict__ out_f,
    int Ntok, int mode, const int* __restrict__ excl_p) {
  __shared__ unsigned short wt[64 * 32];     // 4KB weight K-chunk tile
  __shared__ float cbuf[4][32 * 64];         // 32KB per-wave C tiles
  const int tid = threadIdx.x, lane = tid & 31, wave = tid >> 5;
  const int hi = lane >> 4, ln = lane & 15;
  const int head = blockIdx.y;               // col tile of 64 == head
  const int row0 = blockIdx.x * 128 + wave * 32;

  v8f acc[2][4];
#pragma unroll
  for (int u = 0; u < 2; ++u)
#pragma unroll
    for (int t = 0; t < 4; ++t) acc[u][t] = (v8f){0, 0, 0, 0, 0, 0, 0, 0};

  // staging assignment: thread -> (row r, 32B half of the 64B row)
  const int sr = tid >> 1, sp = tid & 1;
  const unsigned sdst =
      (unsigned)(uintptr_t)(const void*)(wt + sr * 32 + sp * 16);

  for (int k0 = 0; k0 < CDIM; k0 += 32) {
    {  // async stage WT[head*64 .. +63][k0 .. k0+31] into LDS (32B/thread)
      const unsigned short* src =
          WT + (size_t)(head * 64 + sr) * CDIM + k0 + sp * 16;
      async_load_b128(sdst, src);
      async_load_b128(sdst + 16, src + 8);
    }
    wait_asynccnt0();
    __syncthreads();
    v16bf a0 = load_a_frag(X + (size_t)row0 * CDIM + k0, CDIM);
    v16bf a1 = load_a_frag(X + (size_t)(row0 + 16) * CDIM + k0, CDIM);
#pragma unroll
    for (int t = 0; t < 4; ++t) {
      v16bf b = load_b_frag(wt + t * 16 * 32, 32);
      acc[0][t] = wmma_bf16(a0, b, acc[0][t]);
      acc[1][t] = wmma_bf16(a1, b, acc[1][t]);
    }
    __syncthreads();
  }

  if (mode == 3) {  // fp32 output projection, direct coalesced store
#pragma unroll
    for (int u = 0; u < 2; ++u)
#pragma unroll
      for (int t = 0; t < 4; ++t) {
        int col = head * 64 + t * 16 + ln;
        float bv = bias[col];
#pragma unroll
        for (int r = 0; r < 8; ++r)
          out_f[(size_t)(row0 + u * 16 + r + 8 * hi) * CDIM + col] =
              acc[u][t][r] + bv;
      }
    return;
  }

  // wave-private C spill to LDS (same-wave DS ops are in-order; no barrier)
  float* cb = cbuf[wave];
#pragma unroll
  for (int u = 0; u < 2; ++u)
#pragma unroll
    for (int t = 0; t < 4; ++t)
#pragma unroll
      for (int r = 0; r < 8; ++r)
        cb[(u * 16 + r + 8 * hi) * 64 + t * 16 + ln] = acc[u][t][r];

  const int b = row0 / Ntok;   // 2112 % 32 == 0: tiles never cross a batch
  const int n0 = row0 % Ntok;

  if (mode == 2) {  // V: bias + write transposed bf16 [b][h][dv][NKTOT]
#pragma unroll
    for (int t = 0; t < 2; ++t) {
      int dv = lane + 32 * t;
      float bv = bias[head * 64 + dv];
      unsigned int tmp[16];
#pragma unroll
      for (int r = 0; r < 16; ++r)
        tmp[r] = packbf2(cb[(2 * r) * 64 + dv] + bv,
                         cb[(2 * r + 1) * 64 + dv] + bv);
      uint4* dst = (uint4*)(out_h +
          ((size_t)(b * NHEAD + head) * HD + dv) * NKTOT + n0);
#pragma unroll
      for (int r = 0; r < 4; ++r) dst[r] = ((uint4*)tmp)[r];
    }
  } else {  // Q/K: bias + RoPE pair rotation, write [b][h][n][64] bf16
    const int n_keep = (mode == 1) ? (Ntok - excl_p[0]) : Ntok;
    const int i = lane;  // pair index 0..31
    float be = bias[head * 64 + 2 * i];
    float bo2 = bias[head * 64 + 2 * i + 1];
#pragma unroll 4
    for (int r = 0; r < 32; ++r) {
      int n = n0 + r;
      float xe = cb[r * 64 + 2 * i] + be;
      float xo = cb[r * 64 + 2 * i + 1] + bo2;
      float cc = 1.f, ss = 0.f;
      if (n < n_keep && n < NQ) { cc = ropeC[n * 32 + i]; ss = ropeS[n * 32 + i]; }
      float oe = xe * cc - xo * ss;
      float oo = xe * ss + xo * cc;
      ((unsigned int*)out_h)[(((size_t)(b * NHEAD + head) * Ntok + n) * HD +
                              2 * i) >> 1] = packbf2(oe, oo);
    }
  }
}

// ---------------- flash attention (wave-level, bf16 WMMA) ----------------
// All LDS slices are wave-private: no workgroup barriers needed (same-wave DS
// instructions execute in order per ISA, DScnt).
__global__ __launch_bounds__(128) void attn_kernel(
    const unsigned short* __restrict__ qb,   // [B][16][NQ][64]
    const unsigned short* __restrict__ kb,   // [B][16][NKTOT][64]
    const unsigned short* __restrict__ vtb,  // [B][16][64][NKTOT]
    unsigned short* __restrict__ ob) {       // [B][NQ][C] bf16
  __shared__ float Sbuf[4][16 * 32];
  __shared__ unsigned short Pbuf[4][16 * 32];
  __shared__ float Abuf[4][16];
  const int lane = threadIdx.x & 31, wave = threadIdx.x >> 5;
  const int hi = lane >> 4, ln = lane & 15;
  const int row = lane >> 1, half = lane & 1;  // softmax mapping: 2 lanes/row
  const int head = blockIdx.y, b = blockIdx.z;
  const int q0 = blockIdx.x * 64 + wave * 16;
  float* S = Sbuf[wave];
  unsigned short* P = Pbuf[wave];
  float* A = Abuf[wave];

  const unsigned short* qptr = qb + ((size_t)(b * NHEAD + head) * NQ + q0) * HD;
  const unsigned short* kbase = kb + (size_t)(b * NHEAD + head) * NKTOT * HD;
  const unsigned short* vbase = vtb + (size_t)(b * NHEAD + head) * HD * NKTOT;

  v16bf aq0 = load_a_frag(qptr, HD);        // d 0..31
  v16bf aq1 = load_a_frag(qptr + 32, HD);   // d 32..63
  const v8f zero8 = (v8f){0, 0, 0, 0, 0, 0, 0, 0};
  v8f o[4] = {zero8, zero8, zero8, zero8};
  float m_run = -1e30f, l_run = 0.f;        // duplicated across lane pairs

  for (int kc = 0; kc < NKTOT; kc += 32) {
#pragma unroll
    for (int s = 0; s < 2; ++s) {           // two 16-key subtiles
      const unsigned short* kptr = kbase + (size_t)(kc + s * 16) * HD;
      v16bf bk0 = load_b_frag(kptr, HD);        // d 0..31
      v16bf bk1 = load_b_frag(kptr + 32, HD);   // d 32..63
      v8f c = wmma_bf16(aq0, bk0, zero8);
      c = wmma_bf16(aq1, bk1, c);
#pragma unroll
      for (int r = 0; r < 8; ++r)
        S[(r + 8 * hi) * 32 + s * 16 + ln] = c[r] * 0.125f;  // 1/sqrt(64)
    }
    // online softmax: each lane pair handles one q-row (16 keys per lane)
    {
      const float* Srow = S + row * 32 + half * 16;
      float cm = -1e30f;
#pragma unroll
      for (int j = 0; j < 16; ++j) cm = fmaxf(cm, Srow[j]);
      cm = fmaxf(cm, __shfl_xor(cm, 1, 32));
      float nm = fmaxf(m_run, cm);
      float al = __expf(m_run - nm);
      float sum = 0.f;
#pragma unroll
      for (int j = 0; j < 16; j += 2) {
        float p0 = __expf(Srow[j] - nm);
        float p1 = __expf(Srow[j + 1] - nm);
        sum += p0 + p1;
        ((unsigned int*)P)[(row * 32 + half * 16 + j) >> 1] = packbf2(p0, p1);
      }
      sum += __shfl_xor(sum, 1, 32);
      l_run = l_run * al + sum;
      m_run = nm;
      if (half == 0) A[row] = al;
    }
    float al8[8];
#pragma unroll
    for (int r = 0; r < 8; ++r) al8[r] = A[r + 8 * hi];
#pragma unroll
    for (int t = 0; t < 4; ++t)
#pragma unroll
      for (int r = 0; r < 8; ++r) o[t][r] *= al8[r];
    v16bf ap = load_a_frag(P, 32);          // P as A-matrix (16 x 32 keys)
#pragma unroll
    for (int t = 0; t < 4; ++t) {
      v16bf bv = load_b_frag(vbase + (size_t)(t * 16) * NKTOT + kc, NKTOT);
      o[t] = wmma_bf16(ap, bv, o[t]);
    }
  }

  if (half == 0) A[row] = 1.f / l_run;
  float il8[8];
#pragma unroll
  for (int r = 0; r < 8; ++r) il8[r] = A[r + 8 * hi];
#pragma unroll
  for (int t = 0; t < 4; ++t)
#pragma unroll
    for (int r = 0; r < 8; ++r) {
      int q = q0 + r + 8 * hi;
      ob[(size_t)(b * NQ + q) * CDIM + head * 64 + t * 16 + ln] =
          f2bf(o[t][r] * il8[r]);
    }
}

// ---------------- launch ----------------
extern "C" void kernel_launch(void* const* d_in, const int* in_sizes, int n_in,
                              void* d_out, int out_size, void* d_ws,
                              size_t ws_size, hipStream_t stream) {
  (void)in_sizes; (void)n_in; (void)out_size; (void)ws_size;
  const float* q  = (const float*)d_in[0];
  const float* k  = (const float*)d_in[1];
  const float* v  = (const float*)d_in[2];
  const float* Wq = (const float*)d_in[3];
  const float* bq = (const float*)d_in[4];
  const float* Wk = (const float*)d_in[5];
  const float* bk = (const float*)d_in[6];
  const float* Wv = (const float*)d_in[7];
  const float* bv = (const float*)d_in[8];
  const float* Wo = (const float*)d_in[9];
  const float* bo = (const float*)d_in[10];
  const int* qw_p   = (const int*)d_in[12];
  const int* excl_p = (const int*)d_in[13];
  float* out = (float*)d_out;

  char* ws = (char*)d_ws;
  auto take = [&](size_t bytes) -> char* {
    char* p = ws; ws += (bytes + 255) & ~(size_t)255; return p;
  };
  unsigned short* qx  = (unsigned short*)take((size_t)BATCH * NQ * CDIM * 2);
  unsigned short* kx  = (unsigned short*)take((size_t)BATCH * NKTOT * CDIM * 2);
  unsigned short* vx  = (unsigned short*)take((size_t)BATCH * NKTOT * CDIM * 2);
  unsigned short* WqT = (unsigned short*)take((size_t)CDIM * CDIM * 2);
  unsigned short* WkT = (unsigned short*)take((size_t)CDIM * CDIM * 2);
  unsigned short* WvT = (unsigned short*)take((size_t)CDIM * CDIM * 2);
  unsigned short* WoT = (unsigned short*)take((size_t)CDIM * CDIM * 2);
  float* ropeC = (float*)take((size_t)NQ * 32 * 4);
  float* ropeS = (float*)take((size_t)NQ * 32 * 4);
  unsigned short* qbuf = (unsigned short*)take((size_t)BATCH * NHEAD * NQ * HD * 2);
  unsigned short* kbuf = (unsigned short*)take((size_t)BATCH * NHEAD * NKTOT * HD * 2);
  unsigned short* vtb  = (unsigned short*)take((size_t)BATCH * NHEAD * HD * NKTOT * 2);
  unsigned short* obuf = (unsigned short*)take((size_t)BATCH * NQ * CDIM * 2);

  // fp32 -> bf16 copies of activations
  int nq2 = BATCH * NQ * CDIM / 2, nk2 = BATCH * NKTOT * CDIM / 2;
  cvt_bf16_kernel<<<(nq2 + 255) / 256, 256, 0, stream>>>(q, qx, nq2);
  cvt_bf16_kernel<<<(nk2 + 255) / 256, 256, 0, stream>>>(k, kx, nk2);
  cvt_bf16_kernel<<<(nk2 + 255) / 256, 256, 0, stream>>>(v, vx, nk2);

  // transposed bf16 weights
  dim3 tb(32, 8), tg(CDIM / 32, CDIM / 32);
  wt_cvt_kernel<<<tg, tb, 0, stream>>>(Wq, WqT);
  wt_cvt_kernel<<<tg, tb, 0, stream>>>(Wk, WkT);
  wt_cvt_kernel<<<tg, tb, 0, stream>>>(Wv, WvT);
  wt_cvt_kernel<<<tg, tb, 0, stream>>>(Wo, WoT);

  rope_table_kernel<<<(NQ * 32 + 255) / 256, 256, 0, stream>>>(ropeC, ropeS, qw_p);

  // projections (+RoPE / transpose epilogues); block tile = 128 rows
  gemm_bf16_kernel<<<dim3(BATCH * NQ / 128, NHEAD), 128, 0, stream>>>(
      qx, WqT, bq, ropeC, ropeS, qbuf, nullptr, NQ, 0, excl_p);
  gemm_bf16_kernel<<<dim3(BATCH * NKTOT / 128, NHEAD), 128, 0, stream>>>(
      kx, WkT, bk, ropeC, ropeS, kbuf, nullptr, NKTOT, 1, excl_p);
  gemm_bf16_kernel<<<dim3(BATCH * NKTOT / 128, NHEAD), 128, 0, stream>>>(
      vx, WvT, bv, nullptr, nullptr, vtb, nullptr, NKTOT, 2, excl_p);

  // flash attention
  attn_kernel<<<dim3(NQ / 64, NHEAD, BATCH), 128, 0, stream>>>(qbuf, kbuf, vtb,
                                                               obuf);
  // output projection -> fp32 d_out
  gemm_bf16_kernel<<<dim3(BATCH * NQ / 128, NHEAD), 128, 0, stream>>>(
      obuf, WoT, bo, nullptr, nullptr, nullptr, out, NQ, 3, excl_p);
}
